// AttnBlock_29850022708034
// MI455X (gfx1250) — matrix operand order repaired and checked by reading, not compiled, not run on hardware
//
#include <hip/hip_runtime.h>
#include <hip/hip_bf16.h>

// CDNA5 / gfx1250: wave32, WMMA 16x16x32 f16 -> f32, async global->LDS with
// double-buffered pipelining (ASYNCcnt).
typedef __attribute__((ext_vector_type(16))) _Float16 v16h;
typedef __attribute__((ext_vector_type(8)))  _Float16 v8h;
typedef __attribute__((ext_vector_type(8)))  float    v8f;

#define B_      8
#define C_      64
#define HW_     4096
#define GROUPS_ 32
#define EPS_    1e-6f

static __device__ __forceinline__ v8f wmma_f16(v16h a, v16h b, v8f c) {
  // 8 args: (neg_a, A, neg_b, B, c_mod, C, reuse_a, reuse_b)
  return __builtin_amdgcn_wmma_f32_16x16x32_f16(false, a, false, b, (short)0, c,
                                                false, false);
}
static __device__ __forceinline__ v8h ld8(const _Float16* p) {
  return *(const v8h*)p;  // 16B -> ds_load_b128 / global_load_b128
}
static __device__ __forceinline__ v16h cat8(v8h lo, v8h hi) {
  return __builtin_shufflevector(lo, hi, 0, 1, 2, 3, 4, 5, 6, 7, 8, 9, 10, 11,
                                 12, 13, 14, 15);
}
// Async global -> LDS copy of 16 bytes per lane (ASYNCcnt-tracked, no VGPR data)
static __device__ __forceinline__ void async_copy_b128(const _Float16* gsrc,
                                                       _Float16* lds_dst) {
  unsigned lds_off = (unsigned)(size_t)lds_dst;  // low 32b of generic = LDS addr
  asm volatile("global_load_async_to_lds_b128 %0, %1, off" ::"v"(lds_off),
               "v"(gsrc)
               : "memory");
}

// ---------------------------------------------------------------------------
// Kernel 1: GroupNorm statistics. One block per (b, g); reduce 64*64*2 elems.
// ---------------------------------------------------------------------------
__global__ void __launch_bounds__(256) gn_stats_kernel(
    const float* __restrict__ x, float* __restrict__ stats) {
  int bg = blockIdx.x;  // b*32 + g
  int b = bg >> 5, g = bg & 31;
  const float* xb = x + (size_t)b * (HW_ * C_) + g * 2;
  float s = 0.f, ss = 0.f;
  for (int r = threadIdx.x; r < HW_; r += 256) {
    float v0 = xb[(size_t)r * C_ + 0];
    float v1 = xb[(size_t)r * C_ + 1];
    s += v0 + v1;
    ss += v0 * v0 + v1 * v1;
  }
  __shared__ float sh_s[256], sh_q[256];
  sh_s[threadIdx.x] = s;
  sh_q[threadIdx.x] = ss;
  __syncthreads();
  for (int o = 128; o > 0; o >>= 1) {
    if ((int)threadIdx.x < o) {
      sh_s[threadIdx.x] += sh_s[threadIdx.x + o];
      sh_q[threadIdx.x] += sh_q[threadIdx.x + o];
    }
    __syncthreads();
  }
  if (threadIdx.x == 0) {
    float mean = sh_s[0] * (1.f / 8192.f);
    float var = sh_q[0] * (1.f / 8192.f) - mean * mean;
    stats[bg * 2 + 0] = mean;
    stats[bg * 2 + 1] = rsqrtf(var + EPS_);
  }
}

// ---------------------------------------------------------------------------
// Kernel 2: GroupNorm apply + fused Q/K/V projections (WMMA f16).
// Q AND K are stored TRANSPOSED as t[b][j=w*64+d][i=h] so the attention
// kernel's A- and B-fragment loads are contiguous 16B runs. V stays in the
// natural NHWC-flat layout (== the reference's bugged (B,64,4096) view).
// Weights are staged transposed in LDS for contiguous B-fragments here too.
// ---------------------------------------------------------------------------
__global__ void __launch_bounds__(256) gn_qkv_kernel(
    const float* __restrict__ x, const float* __restrict__ scale,
    const float* __restrict__ wq, const float* __restrict__ bq,
    const float* __restrict__ wk, const float* __restrict__ bk,
    const float* __restrict__ wv, const float* __restrict__ bv,
    const float* __restrict__ stats, _Float16* __restrict__ qT,
    _Float16* __restrict__ kT, _Float16* __restrict__ vO) {
  __shared__ _Float16 lds_hn[128][C_ + 8];
  __shared__ _Float16 lds_wT[3][C_][C_ + 8];  // transposed: [d][c]
  __shared__ float lds_bias[3][C_];

  int b = blockIdx.x >> 5;
  int row0 = (blockIdx.x & 31) * 128;  // row within batch (= h*64 + w)
  int tid = threadIdx.x;

  for (int i = tid; i < C_ * C_; i += 256) {
    int c = i >> 6, d = i & 63;  // w*[c][d] row-major
    lds_wT[0][d][c] = (_Float16)wq[i];
    lds_wT[1][d][c] = (_Float16)wk[i];
    lds_wT[2][d][c] = (_Float16)wv[i];
  }
  if (tid < C_) {
    lds_bias[0][tid] = bq[tid];
    lds_bias[1][tid] = bk[tid];
    lds_bias[2][tid] = bv[tid];
  }
  const float* xb = x + (size_t)b * HW_ * C_ + (size_t)row0 * C_;
  for (int i = tid; i < 128 * C_; i += 256) {
    int r = i >> 6, c = i & 63, g = c >> 1;
    float mean = stats[(b * GROUPS_ + g) * 2 + 0];
    float rstd = stats[(b * GROUPS_ + g) * 2 + 1];
    lds_hn[r][c] = (_Float16)((xb[i] - mean) * rstd * scale[c]);
  }
  __syncthreads();

  int wave = tid >> 5, lane = tid & 31;
  int mrow = wave * 16;
  int m = lane & 15;
  int koffA = (lane >> 4) * 8;
  int n = lane & 15;
  int koffB = (lane >> 4) * 8;
  int half_ = lane >> 4;

  // A fragments (16 x 32 f16) for K=64 in 2 steps; contiguous 16B loads.
  const _Float16* ar = &lds_hn[mrow + m][0];
  v16h a0 = cat8(ld8(ar + koffA), ld8(ar + 16 + koffA));
  v16h a1 = cat8(ld8(ar + 32 + koffA), ld8(ar + 48 + koffA));

#pragma unroll
  for (int mat = 0; mat < 3; ++mat) {
#pragma unroll
    for (int nt = 0; nt < 4; ++nt) {
      const _Float16* br = &lds_wT[mat][nt * 16 + n][0];
      v16h b0 = cat8(ld8(br + koffB), ld8(br + 16 + koffB));
      v16h b1 = cat8(ld8(br + 32 + koffB), ld8(br + 48 + koffB));
      v8f acc = {};
      acc = wmma_f16(a0, b0, acc);
      acc = wmma_f16(a1, b1, acc);
#pragma unroll
      for (int r = 0; r < 8; ++r) {
        int rb = row0 + mrow + r + 8 * half_;  // row in batch = h*64 + w
        int d = nt * 16 + n;
        _Float16 hval = (_Float16)(acc[r] + lds_bias[mat][d]);
        int h = rb >> 6, w = rb & 63;
        if (mat == 0) {
          qT[(size_t)b * (C_ * HW_) + (size_t)(w * 64 + d) * 64 + h] = hval;
        } else if (mat == 1) {
          kT[(size_t)b * (C_ * HW_) + (size_t)(w * 64 + d) * 64 + h] = hval;
        } else {
          vO[(size_t)b * (C_ * HW_) + (size_t)rb * C_ + d] = hval;
        }
      }
    }
  }
}

// ---------------------------------------------------------------------------
// Kernel 3: fused flash attention over the bugged layout.
// Per batch: scores = Qt(4096x64) @ K(64x4096), softmax over j2,
// outT[j1][i] = sum_j2 P[j1][j2] * V[i][j2]  (out produced transposed so the
// softmax row stats share the D-fragment row layout).
// K/V 64-wide j2 chunks streamed with DOUBLE-BUFFERED async global->LDS
// copies: chunk c+1's DMA overlaps chunk c's 16 WMMAs + softmax. Async loads
// complete in order, so s_wait_asynccnt 4 == "previous issue fully landed".
// ---------------------------------------------------------------------------
__global__ void __launch_bounds__(256) attn_kernel(
    const _Float16* __restrict__ qT, const _Float16* __restrict__ kT,
    const _Float16* __restrict__ vO, float* __restrict__ outT) {
  __shared__ _Float16 lds_k[2][64][C_ + 8];  // [buf][j2local][i] (K transposed)
  __shared__ _Float16 lds_v[2][C_][64 + 8];  // [buf][i][j2local]
  __shared__ _Float16 lds_p[8][16][64 + 8];

  int b = blockIdx.x >> 5;
  int tid = threadIdx.x;
  int wave = tid >> 5, lane = tid & 31;
  int j1base = (blockIdx.x & 31) * 128 + wave * 16;

  int m = lane & 15;
  int koffA = (lane >> 4) * 8;
  int n = lane & 15;
  int koffB = (lane >> 4) * 8;
  int half_ = lane >> 4;

  const _Float16* kb = kT + (size_t)b * (C_ * HW_);
  const _Float16* vb = vO + (size_t)b * (C_ * HW_);

  // Per-wave: 4 async instructions per chunk issue (2 passes x {K,V}).
  auto issue_chunk = [&](int j2b, int buf) {
#pragma unroll
    for (int pass = 0; pass < 2; ++pass) {
      unsigned f = pass * 2048 + tid * 8;
      unsigned r = f >> 6, c = f & 63;
      // lds_k[buf][r][c] <- kT[b][(j2b+r)*64 + c]  (row r = j2local, col = i)
      async_copy_b128(kb + (size_t)(j2b + r) * 64 + c, &lds_k[buf][r][c]);
      // lds_v[buf][r][c] <- vO[b][r*4096 + j2b + c] (row r = i, col = j2local)
      async_copy_b128(vb + (size_t)r * HW_ + j2b + c, &lds_v[buf][r][c]);
    }
  };

  // Q fragments: A = Qt tile (M=16 j1, K=64 i); qT row-major in (j1, i).
  const _Float16* q = qT + (size_t)b * (C_ * HW_) + (size_t)(j1base + m) * 64;
  v16h aq0 = cat8(ld8(q + koffA), ld8(q + 16 + koffA));
  v16h aq1 = cat8(ld8(q + 32 + koffA), ld8(q + 48 + koffA));

  v8f acc[4] = {v8f{}, v8f{}, v8f{}, v8f{}};  // M=j1(16), N=i(64) as 4 tiles
  float mrow[8], lrow[8];
#pragma unroll
  for (int r = 0; r < 8; ++r) {
    mrow[r] = -1e30f;
    lrow[r] = 0.f;
  }

  issue_chunk(0, 0);  // prologue: chunk 0 in flight

  for (int cidx = 0; cidx < HW_ / 64; ++cidx) {
    int cur = cidx & 1;
    // All waves are done READING buf cur^1 (consumed in the previous iteration).
    __syncthreads();
    if (cidx + 1 < HW_ / 64) {
      issue_chunk((cidx + 1) * 64, cur ^ 1);  // overlap next DMA with compute
      asm volatile("s_wait_asynccnt 4" ::: "memory");  // chunk cidx landed
    } else {
      asm volatile("s_wait_asynccnt 0" ::: "memory");
    }
    __syncthreads();  // chunk cidx visible to every wave

    // scores tile: 16(j1) x 64(j2), contraction K=64 over i.
    v8f sc[4];
#pragma unroll
    for (int t = 0; t < 4; ++t) {
      const _Float16* kr = &lds_k[cur][t * 16 + n][0];
      v16h bk0 = cat8(ld8(kr + koffB), ld8(kr + 16 + koffB));
      v16h bk1 = cat8(ld8(kr + 32 + koffB), ld8(kr + 48 + koffB));
      v8f s = {};
      s = wmma_f16(aq0, bk0, s);
      s = wmma_f16(aq1, bk1, s);
      sc[t] = s;
    }

    // online softmax; row j1 = r + 8*half_ lives across the 16 lanes of a half
    const float scl = 0.125f;  // C^-0.5
#pragma unroll
    for (int r = 0; r < 8; ++r) {
      float vmax = -1e30f;
#pragma unroll
      for (int t = 0; t < 4; ++t) {
        sc[t][r] *= scl;
        vmax = fmaxf(vmax, sc[t][r]);
      }
#pragma unroll
      for (int off = 1; off < 16; off <<= 1)
        vmax = fmaxf(vmax, __shfl_xor(vmax, off, 32));
      float mnew = fmaxf(mrow[r], vmax);
      float corr = __expf(mrow[r] - mnew);
      float psum = 0.f;
#pragma unroll
      for (int t = 0; t < 4; ++t) {
        float p = __expf(sc[t][r] - mnew);
        sc[t][r] = p;
        psum += p;
      }
#pragma unroll
      for (int off = 1; off < 16; off <<= 1) psum += __shfl_xor(psum, off, 32);
      lrow[r] = lrow[r] * corr + psum;
      mrow[r] = mnew;
#pragma unroll
      for (int t = 0; t < 4; ++t) acc[t][r] *= corr;
      int rowm = r + 8 * half_;
#pragma unroll
      for (int t = 0; t < 4; ++t)
        lds_p[wave][rowm][t * 16 + n] = (_Float16)sc[t][r];
    }
    __builtin_amdgcn_wave_barrier();  // per-wave LDS round-trip (DS in-order)

    // PV: acc(16 j1 x 64 i) += P(16 x 64 j2) @ V^T(j2 x i), K=32 per step.
#pragma unroll
    for (int ks = 0; ks < 2; ++ks) {
      const _Float16* pr = &lds_p[wave][m][ks * 32];
      v16h ap = cat8(ld8(pr + koffA), ld8(pr + 16 + koffA));
#pragma unroll
      for (int t = 0; t < 4; ++t) {
        const _Float16* vr = &lds_v[cur][t * 16 + n][ks * 32];
        v16h bv = cat8(ld8(vr + koffB), ld8(vr + 16 + koffB));
        acc[t] = wmma_f16(ap, bv, acc[t]);
      }
    }
  }

  // normalize and store outT[b][j1][i] (f32)
#pragma unroll
  for (int r = 0; r < 8; ++r) {
    float inv = 1.0f / lrow[r];
    int j1 = j1base + r + 8 * half_;
#pragma unroll
    for (int t = 0; t < 4; ++t)
      outT[(size_t)b * (HW_ * C_) + (size_t)j1 * C_ + t * 16 + n] =
          acc[t][r] * inv;
  }
}

// ---------------------------------------------------------------------------
// Kernel 4: out-projection over the last (w) axis + bias + flat residual.
// One block per (b, h): A[c][w] = outT[b][h*64+w][c]; out = A @ wp + bp + x.
// ---------------------------------------------------------------------------
__global__ void __launch_bounds__(256) proj_kernel(
    const float* __restrict__ outT, const float* __restrict__ wp,
    const float* __restrict__ bp, const float* __restrict__ x,
    float* __restrict__ out) {
  __shared__ _Float16 lds_a[C_][C_ + 8];    // [c][w]
  __shared__ _Float16 lds_wpT[C_][C_ + 8];  // transposed: [d][w]
  __shared__ float lds_bp[C_];

  int bh = blockIdx.x;
  int b = bh >> 6, h = bh & 63;
  int tid = threadIdx.x;

  const float* src = outT + (size_t)b * (HW_ * C_) + (size_t)h * 4096;
  for (int i = tid; i < 64 * 64; i += 256) {
    int w = i >> 6, c = i & 63;
    lds_a[c][w] = (_Float16)src[i];          // coalesced read, LDS transpose
    lds_wpT[i & 63][i >> 6] = (_Float16)wp[i];
  }
  if (tid < C_) lds_bp[tid] = bp[tid];
  __syncthreads();

  int wave = tid >> 5, lane = tid & 31;
  int m = lane & 15;
  int koffA = (lane >> 4) * 8;
  int n = lane & 15;
  int koffB = (lane >> 4) * 8;
  int half_ = lane >> 4;

#pragma unroll
  for (int s = 0; s < 2; ++s) {
    int t8 = wave * 2 + s;  // 16 tiles of the 64x64 output
    int mt = t8 >> 2, nt = t8 & 3;
    const _Float16* ar = &lds_a[mt * 16 + m][0];
    v16h a0 = cat8(ld8(ar + koffA), ld8(ar + 16 + koffA));
    v16h a1 = cat8(ld8(ar + 32 + koffA), ld8(ar + 48 + koffA));
    const _Float16* br = &lds_wpT[nt * 16 + n][0];
    v16h b0 = cat8(ld8(br + koffB), ld8(br + 16 + koffB));
    v16h b1 = cat8(ld8(br + 32 + koffB), ld8(br + 48 + koffB));
    v8f acc = {};
    acc = wmma_f16(a0, b0, acc);
    acc = wmma_f16(a1, b1, acc);
#pragma unroll
    for (int r = 0; r < 8; ++r) {
      int cc = mt * 16 + r + 8 * half_;
      int d = nt * 16 + n;
      size_t idx = (size_t)b * (HW_ * C_) + (size_t)cc * 4096 + h * 64 + d;
      out[idx] = acc[r] + lds_bp[d] + x[idx];
    }
  }
}

// ---------------------------------------------------------------------------
extern "C" void kernel_launch(void* const* d_in, const int* in_sizes, int n_in,
                              void* d_out, int out_size, void* d_ws,
                              size_t ws_size, hipStream_t stream) {
  (void)in_sizes;
  (void)n_in;
  (void)out_size;
  (void)ws_size;
  const float* x = (const float*)d_in[0];
  const float* nsc = (const float*)d_in[1];
  const float* wq = (const float*)d_in[2];
  const float* bq = (const float*)d_in[3];
  const float* wk = (const float*)d_in[4];
  const float* bk = (const float*)d_in[5];
  const float* wv = (const float*)d_in[6];
  const float* bv = (const float*)d_in[7];
  const float* wp = (const float*)d_in[8];
  const float* bp = (const float*)d_in[9];
  float* out = (float*)d_out;

  // Workspace layout (bytes): ~20 MB total.
  char* ws = (char*)d_ws;
  float* stats = (float*)ws;                            //  2 KB
  _Float16* qT = (_Float16*)(ws + 4096);                //  4 MB
  _Float16* kT = (_Float16*)(ws + 4096 + 4194304);      //  4 MB
  _Float16* vO = (_Float16*)(ws + 4096 + 2 * 4194304);  //  4 MB
  float* outT = (float*)(ws + 4096 + 3 * 4194304);      //  8 MB

  gn_stats_kernel<<<B_ * GROUPS_, 256, 0, stream>>>(x, stats);
  gn_qkv_kernel<<<B_ * 32, 256, 0, stream>>>(x, nsc, wq, bq, wk, bk, wv, bv,
                                             stats, qT, kT, vO);
  attn_kernel<<<B_ * 32, 256, 0, stream>>>(qT, kT, vO, outT);
  proj_kernel<<<B_ * C_, 256, 0, stream>>>(outT, wp, bp, x, out);
}
